// EGNN_Triton_Layer_36713380446826
// MI455X (gfx1250) — matrix operand order repaired and checked by reading, not compiled, not run on hardware
//
#include <hip/hip_runtime.h>
#include <hip/hip_bf16.h>

#define N_NODES_C 50000
#define N_EDGES_C 1600000

typedef __attribute__((ext_vector_type(16))) _Float16 v16h;
typedef __attribute__((ext_vector_type(8)))  _Float16 v8h;
typedef __attribute__((ext_vector_type(8)))  float    v8f;

union H16 { v16h v; v8h h[2]; _Float16 e[16]; };
union F8  { v8f v; float e[8]; };

// ---- workspace layout (f16 element offsets unless noted) ----
#define W1M_OFF 0        // 6*8*512  = 24576 f16  (w1_msg, K=176->192, N=128)
#define W2M_OFF 24576    // 4*4*512  =  8192 f16  (w2_msg, K=128, N=64)
#define WMV_OFF 32768    // 2*4*512  =  4096 f16  (w1_mov, K=64,  N=64)
#define W1N_OFF 36864    // 4*8*512  = 16384 f16  (w1_node,K=128, N=128)
#define W2N_OFF 53248    // 4*4*512  =  8192 f16  (w2_node,K=128, N=64)
#define TOTMSG_BYTE 131072  // f32 total_msg[50000*64] after 128KB of weights

// fast SiLU: x * rcp(1+exp(-x)) -- v_exp_f32 + v_rcp_f32, no IEEE div chain
__device__ __forceinline__ float silu_f(float x) {
  return x * __builtin_amdgcn_rcpf(1.0f + __expf(-x));
}

__device__ __forceinline__ v8f wmma_f16(v16h a, v16h b, v8f c) {
  return __builtin_amdgcn_wmma_f32_16x16x32_f16(false, a, false, b, (short)0, c, false, false);
}

// A-fragment (16x32 f16, ISA 7.12.2): lane row = lane&15, half g = lane>>4
// lane holds K = kk*32 + {g*8..g*8+7} and {16+g*8 .. 16+g*8+7}
__device__ __forceinline__ v16h lds_afrag(const _Float16* tile, int rowStride, int kk, int lane) {
  int g = lane >> 4, row = lane & 15;
  const _Float16* p = tile + row * rowStride + kk * 32 + g * 8;
  H16 u;
  u.h[0] = *(const v8h*)(p);
  u.h[1] = *(const v8h*)(p + 16);
  return u.v;
}

// B-fragment pre-swizzled in workspace: frag (kk,nt) is 512 contiguous f16,
// lane chunk = 16 contiguous f16 (32B aligned)
__device__ __forceinline__ v16h g_bfrag(const _Float16* frags, int ntiles, int kk, int nt, int lane) {
  return *(const v16h*)(frags + (size_t)((kk * ntiles + nt) * 512 + lane * 16));
}

__device__ __forceinline__ void atomic_add_global(float* p, float v) {
  __hip_atomic_fetch_add(p, v, __ATOMIC_RELAXED, __HIP_MEMORY_SCOPE_AGENT);
}
__device__ __forceinline__ void atomic_add_lds(float* p, float v) {
  __hip_atomic_fetch_add(p, v, __ATOMIC_RELAXED, __HIP_MEMORY_SCOPE_WORKGROUP);
}

// ---------------- prep: zero accumulator, init out coords, swizzle weights to f16 B-frags ----
__device__ __forceinline__ void swz(const float* W, _Float16* dst, int idx, int Ksrc, int N) {
  int ntiles = N >> 4;
  int kt  = idx / (ntiles * 512);
  int rem = idx - kt * (ntiles * 512);
  int nt  = rem >> 9;
  int l   = rem & 511;
  int lane = l >> 4, p = l & 15;
  int k = kt * 32 + ((p >> 3) << 4) + ((lane >> 4) << 3) + (p & 7);
  int n = nt * 16 + (lane & 15);
  float v = (k < Ksrc) ? W[k * N + n] : 0.0f;
  dst[idx] = (_Float16)v;
}

__global__ void prep_kernel(const float* __restrict__ coord,
                            const float* __restrict__ w1_msg,
                            const float* __restrict__ w2_msg,
                            const float* __restrict__ w1_mov,
                            const float* __restrict__ w1_node,
                            const float* __restrict__ w2_node,
                            _Float16* __restrict__ wsf16,
                            float* __restrict__ total_msg,
                            float* __restrict__ out_coord) {
  int i = blockIdx.x * blockDim.x + threadIdx.x;
  if (i < N_NODES_C * 64) { total_msg[i] = 0.0f; return; }
  i -= N_NODES_C * 64;
  if (i < N_NODES_C * 3) { out_coord[i] = coord[i]; return; }
  i -= N_NODES_C * 3;
  if (i < 24576) { swz(w1_msg,  wsf16 + W1M_OFF, i, 176, 128); return; }
  i -= 24576;
  if (i < 8192)  { swz(w2_msg,  wsf16 + W2M_OFF, i, 128, 64);  return; }
  i -= 8192;
  if (i < 4096)  { swz(w1_mov,  wsf16 + WMV_OFF, i, 64, 64);   return; }
  i -= 4096;
  if (i < 16384) { swz(w1_node, wsf16 + W1N_OFF, i, 128, 128); return; }
  i -= 16384;
  if (i < 8192)  { swz(w2_node, wsf16 + W2N_OFF, i, 128, 64);  return; }
}

// ---------------- edge kernel: 1 wave = 16 edges, WMMA chain 192->128->64->64->scalar ----
#define EW 4  // waves per block

__global__ __launch_bounds__(128) void edge_kernel(
    const float* __restrict__ node_feat, const float* __restrict__ coord,
    const float* __restrict__ edge_feat, const int* __restrict__ edge_index,
    const float* __restrict__ rbf_centers, const float* __restrict__ w2_mov,
    const _Float16* __restrict__ wsf16,
    float* __restrict__ total_msg, float* __restrict__ out_coord) {
  __shared__ _Float16 s_min[EW][16 * 192];
  __shared__ _Float16 s_h[EW][16 * 128];
  __shared__ float s_msg[EW][16 * 64];
  __shared__ float s_rel[EW][16 * 3];
  __shared__ float s_dist[EW][16];
  __shared__ int   s_src[EW][16];
  __shared__ int   s_dst[EW][16];
  __shared__ float s_s[EW][16];

  const int w    = threadIdx.x >> 5;
  const int lane = threadIdx.x & 31;
  const int tile = blockIdx.x * EW + w;   // 100000 tiles exactly, no guard
  const int e0   = tile * 16;
  const int g    = lane >> 4;
  const int cI   = lane & 15;

  // geometry + indices (lanes 0..15, one edge each)
  if (lane < 16) {
    int e  = e0 + lane;
    int si = edge_index[e];
    int di = edge_index[N_EDGES_C + e];
    s_src[w][lane] = si;
    s_dst[w][lane] = di;
    float rx = coord[di * 3 + 0] - coord[si * 3 + 0];
    float ry = coord[di * 3 + 1] - coord[si * 3 + 1];
    float rz = coord[di * 3 + 2] - coord[si * 3 + 2];
    s_rel[w][lane * 3 + 0] = rx;
    s_rel[w][lane * 3 + 1] = ry;
    s_rel[w][lane * 3 + 2] = rz;
    s_dist[w][lane] = sqrtf(rx * rx + ry * ry + rz * rz);
    s_s[w][lane] = 0.0f;
  }
  __syncthreads();

  // build m_in tile [16 x 192] f16: [src_feat 64 | dst_feat 64 | rbf 32 | edge_feat 16 | pad 16]
  for (int i = lane; i < 16 * 192; i += 32) {
    int row = i / 192, col = i - row * 192;
    float v;
    if (col < 64)        v = node_feat[(size_t)s_src[w][row] * 64 + col];
    else if (col < 128)  v = node_feat[(size_t)s_dst[w][row] * 64 + (col - 64)];
    else if (col < 160) { float d = s_dist[w][row] - rbf_centers[col - 128];
                          v = __expf(-10.0f * d * d); }
    else if (col < 176)  v = edge_feat[(size_t)(e0 + row) * 16 + (col - 160)];
    else                 v = 0.0f;
    s_min[w][i] = (_Float16)v;
  }
  __syncthreads();

  // ---- layer 1: h = silu(m_in @ W1) : K=192 (6 steps), N=128 (8 tiles)
  v16h a1[6];
  for (int kk = 0; kk < 6; ++kk) a1[kk] = lds_afrag(&s_min[w][0], 192, kk, lane);
  for (int nt = 0; nt < 8; ++nt) {
    v8f acc = {};
    for (int kk = 0; kk < 6; ++kk)
      acc = wmma_f16(a1[kk], g_bfrag(wsf16 + W1M_OFF, 8, kk, nt, lane), acc);
    F8 u; u.v = acc;
    for (int r = 0; r < 8; ++r) {
      int row = r + g * 8;
      s_h[w][row * 128 + nt * 16 + cI] = (_Float16)silu_f(u.e[r]);
    }
  }
  __syncthreads();

  // ---- layer 2: msg = silu(h @ W2) : K=128 (4 steps), N=64 (4 tiles); scatter-add to nodes
  v16h a2[4];
  for (int kk = 0; kk < 4; ++kk) a2[kk] = lds_afrag(&s_h[w][0], 128, kk, lane);
  for (int nt = 0; nt < 4; ++nt) {
    v8f acc = {};
    for (int kk = 0; kk < 4; ++kk)
      acc = wmma_f16(a2[kk], g_bfrag(wsf16 + W2M_OFF, 4, kk, nt, lane), acc);
    F8 u; u.v = acc;
    for (int r = 0; r < 8; ++r) {
      int row = r + g * 8;
      int c   = nt * 16 + cI;
      float v = silu_f(u.e[r]);
      s_msg[w][row * 64 + c] = v;
      atomic_add_global(&total_msg[(size_t)s_dst[w][row] * 64 + c], v);
    }
  }
  __syncthreads();

  // ---- layer 3: t = silu(msg @ Wmov), s = t @ w2_mov : K=64 (2 steps), N=64 (4 tiles)
  v16h a3[2];
  {
    int row = lane & 15;
    for (int kk = 0; kk < 2; ++kk) {
      const float* p = &s_msg[w][row * 64 + kk * 32];
      H16 u;
      for (int t = 0; t < 8; ++t) u.e[t]     = (_Float16)p[g * 8 + t];
      for (int t = 0; t < 8; ++t) u.e[8 + t] = (_Float16)p[16 + g * 8 + t];
      a3[kk] = u.v;
    }
  }
  for (int nt = 0; nt < 4; ++nt) {
    v8f acc = {};
    for (int kk = 0; kk < 2; ++kk)
      acc = wmma_f16(a3[kk], g_bfrag(wsf16 + WMV_OFF, 4, kk, nt, lane), acc);
    F8 u; u.v = acc;
    float wv = w2_mov[nt * 16 + cI];
    for (int r = 0; r < 8; ++r)
      atomic_add_lds(&s_s[w][r + g * 8], silu_f(u.e[r]) * wv);
  }
  __syncthreads();

  // mov = rel * s -> atomic add to output coords
  if (lane < 16) {
    float sv = s_s[w][lane];
    int di = s_dst[w][lane];
    atomic_add_global(&out_coord[di * 3 + 0], s_rel[w][lane * 3 + 0] * sv);
    atomic_add_global(&out_coord[di * 3 + 1], s_rel[w][lane * 3 + 1] * sv);
    atomic_add_global(&out_coord[di * 3 + 2], s_rel[w][lane * 3 + 2] * sv);
  }
}

// ---------------- node kernel: 1 wave = 16 nodes, [16x128] -> 128 -> 64 + residual ----
__global__ __launch_bounds__(128) void node_kernel(
    const float* __restrict__ node_feat,
    const float* __restrict__ b1, const float* __restrict__ b2,
    const _Float16* __restrict__ wsf16, const float* __restrict__ total_msg,
    float* __restrict__ out_feat) {
  __shared__ _Float16 s_in[EW][16 * 128];
  __shared__ _Float16 s_h[EW][16 * 128];

  const int w    = threadIdx.x >> 5;
  const int lane = threadIdx.x & 31;
  int tile = blockIdx.x * EW + w;
  const int NT = N_NODES_C / 16;          // 3125
  if (tile >= NT) tile = NT - 1;          // duplicate last tile (identical writes, no atomics)
  const int n0 = tile * 16;
  const int g  = lane >> 4;
  const int cI = lane & 15;

  for (int i = lane; i < 16 * 128; i += 32) {
    int row = i >> 7, col = i & 127;
    float v = (col < 64) ? node_feat[(size_t)(n0 + row) * 64 + col]
                         : total_msg[(size_t)(n0 + row) * 64 + (col - 64)];
    s_in[w][i] = (_Float16)v;
  }
  __syncthreads();

  v16h a[4];
  for (int kk = 0; kk < 4; ++kk) a[kk] = lds_afrag(&s_in[w][0], 128, kk, lane);
  for (int nt = 0; nt < 8; ++nt) {
    v8f acc = {};
    for (int kk = 0; kk < 4; ++kk)
      acc = wmma_f16(a[kk], g_bfrag(wsf16 + W1N_OFF, 8, kk, nt, lane), acc);
    F8 u; u.v = acc;
    float bias = b1[nt * 16 + cI];
    for (int r = 0; r < 8; ++r) {
      int row = r + g * 8;
      s_h[w][row * 128 + nt * 16 + cI] = (_Float16)silu_f(u.e[r] + bias);
    }
  }
  __syncthreads();

  v16h a2[4];
  for (int kk = 0; kk < 4; ++kk) a2[kk] = lds_afrag(&s_h[w][0], 128, kk, lane);
  for (int nt = 0; nt < 4; ++nt) {
    v8f acc = {};
    for (int kk = 0; kk < 4; ++kk)
      acc = wmma_f16(a2[kk], g_bfrag(wsf16 + W2N_OFF, 4, kk, nt, lane), acc);
    F8 u; u.v = acc;
    float bias = b2[nt * 16 + cI];
    for (int r = 0; r < 8; ++r) {
      int row = r + g * 8;
      size_t idx = (size_t)(n0 + row) * 64 + nt * 16 + cI;
      out_feat[idx] = node_feat[idx] + u.e[r] + bias;
    }
  }
}

extern "C" void kernel_launch(void* const* d_in, const int* in_sizes, int n_in,
                              void* d_out, int out_size, void* d_ws, size_t ws_size,
                              hipStream_t stream) {
  const float* node_feat   = (const float*)d_in[0];
  const float* coord       = (const float*)d_in[1];
  const float* edge_feat   = (const float*)d_in[2];
  const int*   edge_index  = (const int*)d_in[3];
  const float* w1_msg      = (const float*)d_in[4];
  const float* w2_msg      = (const float*)d_in[5];
  const float* w1_mov      = (const float*)d_in[6];
  const float* w2_mov      = (const float*)d_in[7];
  const float* rbf_centers = (const float*)d_in[8];
  const float* w1_node     = (const float*)d_in[9];
  const float* b1_node     = (const float*)d_in[10];
  const float* w2_node     = (const float*)d_in[11];
  const float* b2_node     = (const float*)d_in[12];

  float* out       = (float*)d_out;
  float* out_feat  = out;
  float* out_coord = out + (size_t)N_NODES_C * 64;

  _Float16* wsf16  = (_Float16*)d_ws;
  float* total_msg = (float*)((char*)d_ws + TOTMSG_BYTE);

  // prep: 3,200,000 (zero) + 150,000 (coord copy) + 61,440 (swizzle) threads
  const int prep_total = N_NODES_C * 64 + N_NODES_C * 3 + 24576 + 8192 + 4096 + 16384 + 8192;
  prep_kernel<<<(prep_total + 255) / 256, 256, 0, stream>>>(
      coord, w1_msg, w2_msg, w1_mov, w1_node, w2_node, wsf16, total_msg, out_coord);

  // edges: 1.6M/16 = 100000 tiles / 4 waves = 25000 blocks
  edge_kernel<<<25000, 128, 0, stream>>>(
      node_feat, coord, edge_feat, edge_index, rbf_centers, w2_mov,
      wsf16, total_msg, out_coord);

  // nodes: 3125 tiles / 4 waves -> 782 blocks (last tiles clamp)
  node_kernel<<<(N_NODES_C / 16 + EW - 1) / EW, 128, 0, stream>>>(
      node_feat, b1_node, b2_node, wsf16, total_msg, out_feat);
}
